// LinearAttention_32375463477924
// MI455X (gfx1250) — matrix operand order repaired and checked by reading, not compiled, not run on hardware
//
#include <hip/hip_runtime.h>
#include <hip/hip_bf16.h>

typedef __bf16 bf16;
typedef __attribute__((ext_vector_type(16))) __bf16 v16bf;
typedef __attribute__((ext_vector_type(8)))  __bf16 v8bf;
typedef __attribute__((ext_vector_type(8)))  float   v8f;

#define DEV static __device__ __forceinline__

// ---- constants from the reference ----
#define BATCH   4
#define SEQ     8192
#define DIMM    1024
#define HEADS   16
#define DHEAD   64
#define INNER   1024
#define TRIPLE  3072
#define MROWS   (BATCH * SEQ)      // 32768
#define EPSV    1e-6f

// LDS strides (bf16 elements); all row byte-strides are multiples of 16B.
#define LDA_S  40    // GEMM A tile 128x32 (+8 pad)   -> 80B rows
#define LDBT   40    // GEMM Bt tile 128(n)x32(k)(+8) -> 80B rows
#define KVN_S  136   // kv-kernel K/V tiles 64(d)x128(n)(+8) -> 272B rows
#define KVT_S  72    // attn kvT 64(f)x64(d)(+8)      -> 144B rows

DEV float elu1(float v) { return v > 0.f ? v + 1.f : __expf(v); }

DEV v8f zero8() {
    v8f z;
#pragma unroll
    for (int i = 0; i < 8; ++i) z[i] = 0.f;
    return z;
}

// ---- CDNA5 async global->LDS copy (per-lane 16B), ASYNCcnt tracked ----
DEV void async_ld_b128(bf16* lds, const bf16* gsrc) {
    asm volatile("global_load_async_to_lds_b128 %0, %1, off"
                 :: "v"((unsigned)(unsigned long long)lds), "v"(gsrc)
                 : "memory");
}
DEV void wait_async0() { asm volatile("s_wait_asynccnt 0x0" ::: "memory"); }

// ---------------- WMMA fragment builders (ISA 7.12.2 layouts) ----------------
// A fragment from a row-major [16 x 32] bf16 tile: two contiguous 16B runs.
DEV v16bf frag_a(const bf16* p, int stride) {
    const int lane = threadIdx.x & 31;
    const int row  = lane & 15;
    const int kb   = (lane >> 4) << 3;          // 0 or 8
    const bf16* r  = p + row * stride;
    v16bf a;
#pragma unroll
    for (int i = 0; i < 8; ++i) {
        a[i]     = r[kb + i];
        a[i + 8] = r[16 + kb + i];
    }
    return a;
}

// B fragment from a TRANSPOSED [16(n) x 32(k)] tile: two contiguous 8-k runs per lane.
DEV v16bf frag_bT(const bf16* p, int stride) {
    const int lane = threadIdx.x & 31;
    const int col  = lane & 15;
    const int kb   = (lane >> 4) << 3;          // 0 or 8
    const bf16* r  = p + col * stride + kb;
    v16bf b;
#pragma unroll
    for (int i = 0; i < 8; ++i) { b[i] = r[i]; b[i + 8] = r[16 + i]; }
    return b;
}

#define WMMA_BF16(acc, a, b)                                                     \
    acc = __builtin_amdgcn_wmma_f32_16x16x32_bf16(false, (a), false, (b),        \
                                                  (short)0, (acc), false, false)

// ---------------------------------------------------------------------------
// Weight convert + transpose: S[rows][cols] f32  ->  D[cols][rows] bf16
// ---------------------------------------------------------------------------
__global__ __launch_bounds__(256) void cvt_transpose(
    const float* __restrict__ S, bf16* __restrict__ D, int rows, int cols)
{
    __shared__ float tile[32][33];
    const int bx = blockIdx.x * 32;
    const int by = blockIdx.y * 32;
    const int lx = threadIdx.x & 31;
    const int ly = threadIdx.x >> 5;
#pragma unroll
    for (int i = 0; i < 32; i += 8)
        tile[ly + i][lx] = S[(size_t)(by + ly + i) * cols + bx + lx];
    __syncthreads();
#pragma unroll
    for (int i = 0; i < 32; i += 8)
        D[(size_t)(bx + ly + i) * rows + by + lx] = (bf16)tile[lx][ly + i];
}

__global__ void zero_f32(float* __restrict__ p, int n) {
    int i = blockIdx.x * blockDim.x + threadIdx.x;
    if (i < n) p[i] = 0.f;
}

// ---------------------------------------------------------------------------
// GEMM1: qkv = x @ Wqkv. Double-buffered: B tile via async-to-LDS, A tile via
// register prefetch + v_cvt_pk_bf16_f32. Wave tile 64(M)x32(N).
// ---------------------------------------------------------------------------
__global__ __launch_bounds__(256) void gemm_qkv(
    const float* __restrict__ X,       // [32768,1024]
    const bf16*  __restrict__ WT,      // [3072,1024] bf16 (transposed)
    bf16* __restrict__ Q,              // [bh][n][d]
    bf16* __restrict__ KT,             // [bh][d][n]
    bf16* __restrict__ VT)             // [bh][d][n]
{
    __shared__ bf16 As[2][128 * LDA_S];
    __shared__ bf16 Bt[2][128 * LDBT];

    const int tid     = threadIdx.x;
    const int wave    = tid >> 5;
    const int lane    = tid & 31;
    const int rowBase = blockIdx.y * 128;
    const int colBase = blockIdx.x * 128;
    const int wm      = (wave >> 2) * 64;     // 0 or 64
    const int wn      = (wave & 3) * 32;      // 0,32,64,96

    v8f acc[4][2];
#pragma unroll
    for (int i = 0; i < 4; ++i)
#pragma unroll
        for (int j = 0; j < 2; ++j) acc[i][j] = zero8();

    // staging map: 256 threads, 128 rows x 32 cols, 16 elems/thread
    const int ar = tid >> 1;
    const int ac = (tid & 1) << 4;
    const float* xrow = X  + (size_t)(rowBase + ar) * DIMM + ac;
    const bf16*  wrow = WT + (size_t)(colBase + ar) * DIMM + ac;

    float4 fa[4];
    auto loadA = [&](int k0) {
        const float4* p = (const float4*)(xrow + k0);
        fa[0] = p[0]; fa[1] = p[1]; fa[2] = p[2]; fa[3] = p[3];
    };
    auto storeA = [&](bf16* dst) {
        v8bf h0, h1;
        h0[0]=(bf16)fa[0].x; h0[1]=(bf16)fa[0].y; h0[2]=(bf16)fa[0].z; h0[3]=(bf16)fa[0].w;
        h0[4]=(bf16)fa[1].x; h0[5]=(bf16)fa[1].y; h0[6]=(bf16)fa[1].z; h0[7]=(bf16)fa[1].w;
        h1[0]=(bf16)fa[2].x; h1[1]=(bf16)fa[2].y; h1[2]=(bf16)fa[2].z; h1[3]=(bf16)fa[2].w;
        h1[4]=(bf16)fa[3].x; h1[5]=(bf16)fa[3].y; h1[6]=(bf16)fa[3].z; h1[7]=(bf16)fa[3].w;
        bf16* a = dst + ar * LDA_S + ac;
        *(v8bf*)a       = h0;
        *(v8bf*)(a + 8) = h1;
    };
    auto asyncB = [&](int buf, int k0) {
        bf16* dst = &Bt[buf][0] + ar * LDBT + ac;
        const bf16* src = wrow + k0;
        async_ld_b128(dst,     src);
        async_ld_b128(dst + 8, src + 8);
    };
    auto compute = [&](const bf16* Ab, const bf16* Bb) {
        v16bf bfr[2];
#pragma unroll
        for (int tj = 0; tj < 2; ++tj)
            bfr[tj] = frag_bT(Bb + (wn + tj * 16) * LDBT, LDBT);
#pragma unroll
        for (int ti = 0; ti < 4; ++ti) {
            v16bf a = frag_a(Ab + (wm + ti * 16) * LDA_S, LDA_S);
            WMMA_BF16(acc[ti][0], a, bfr[0]);
            WMMA_BF16(acc[ti][1], a, bfr[1]);
        }
    };

    loadA(0);
    asyncB(0, 0);
    storeA(&As[0][0]);
    int cur = 0;
    for (int k0 = 0; k0 < DIMM; k0 += 32) {
        const bool more = (k0 + 32) < DIMM;
        if (more) loadA(k0 + 32);
        wait_async0();              // our async writes into Bt[cur] complete
        __syncthreads();            // everyone done writing cur / reading cur^1
        if (more) asyncB(cur ^ 1, k0 + 32);
        compute(&As[cur][0], &Bt[cur][0]);
        if (more) { storeA(&As[cur ^ 1][0]); cur ^= 1; }
    }

    // ---- epilogue: `which` is block-uniform (128 | 1024) ----
    const int which  = colBase >> 10;          // 0:q 1:k 2:v
    const int cmBase = (colBase & 1023) + wn;
    const int lrow   = (lane >> 4) << 3;
    const int lcol   = lane & 15;

    if (which == 0) {
#pragma unroll
        for (int ti = 0; ti < 4; ++ti)
#pragma unroll
            for (int tj = 0; tj < 2; ++tj) {
                const int c = cmBase + tj * 16 + lcol;
                const int h = c >> 6, d = c & 63;
#pragma unroll
                for (int r = 0; r < 8; ++r) {
                    const int m = rowBase + wm + ti * 16 + lrow + r;
                    const int b = m >> 13, n = m & 8191;
                    Q[(((size_t)b * HEADS + h) * SEQ + n) * DHEAD + d] =
                        (bf16)elu1(acc[ti][tj][r]);
                }
            }
    } else {
        bf16* T = (which == 1) ? KT : VT;
        const bool doElu = (which == 1);
#pragma unroll
        for (int ti = 0; ti < 4; ++ti)
#pragma unroll
            for (int tj = 0; tj < 2; ++tj) {
                const int c = cmBase + tj * 16 + lcol;
                const int h = c >> 6, d = c & 63;
                const int m0 = rowBase + wm + ti * 16 + lrow;  // 8 consecutive n
                const int b = m0 >> 13, nn = m0 & 8191;
                v8bf pk;
#pragma unroll
                for (int r = 0; r < 8; ++r) {
                    float v = acc[ti][tj][r];
                    pk[r] = (bf16)(doElu ? elu1(v) : v);
                }
                *(v8bf*)(T + (((size_t)b * HEADS + h) * DHEAD + d) * SEQ + nn) = pk;
            }
    }
}

// ---------------------------------------------------------------------------
// ksum[bh][d] = sum_n K (coalesced row reduction over KT[bh][d][n])
// ---------------------------------------------------------------------------
__global__ __launch_bounds__(256) void ksum_kernel(
    const bf16* __restrict__ KT, float* __restrict__ Ksum)
{
    const int tid = threadIdx.x;
    const v8bf* rv = (const v8bf*)(KT + (size_t)blockIdx.x * SEQ);
    float s = 0.f;
#pragma unroll
    for (int it = 0; it < 4; ++it) {
        v8bf q = rv[tid + it * 256];
#pragma unroll
        for (int j = 0; j < 8; ++j) s += (float)q[j];
    }
    s += __shfl_xor(s, 16, 32);
    s += __shfl_xor(s,  8, 32);
    s += __shfl_xor(s,  4, 32);
    s += __shfl_xor(s,  2, 32);
    s += __shfl_xor(s,  1, 32);
    __shared__ float red[8];
    if ((tid & 31) == 0) red[tid >> 5] = s;
    __syncthreads();
    if (tid == 0) {
        float t = 0.f;
#pragma unroll
        for (int w = 0; w < 8; ++w) t += red[w];
        Ksum[blockIdx.x] = t;
    }
}

// ---------------------------------------------------------------------------
// kv[bh] = K^T V (64x64, f32 atomic reduce over 4 seq segments).
// Double-buffered async staging; 512 thr = 16 waves, one 16x16 tile each.
// ---------------------------------------------------------------------------
__global__ __launch_bounds__(512) void kv_kernel(
    const bf16* __restrict__ KT, const bf16* __restrict__ VT,
    float* __restrict__ KV)
{
    __shared__ bf16 Ks[2][DHEAD * KVN_S];
    __shared__ bf16 Vs[2][DHEAD * KVN_S];

    const int tid  = threadIdx.x;
    const int bh   = blockIdx.x;
    const int seg  = blockIdx.y;
    const int wave = tid >> 5;
    const int lane = tid & 31;
    const int ti   = (wave >> 2) * 16;   // d-row tile of kv
    const int tj   = (wave & 3) * 16;    // f-col tile of kv

    const size_t base = (size_t)bh * DHEAD * SEQ + (size_t)seg * 2048;
    const int ld = tid >> 3;             // d row 0..63
    const int ln = (tid & 7) << 4;       // n col 0..112
    const bf16* kp0 = KT + base + (size_t)ld * SEQ + ln;
    const bf16* vp0 = VT + base + (size_t)ld * SEQ + ln;

    auto asyncKV = [&](int buf, int n0) {
        bf16* kd = &Ks[buf][0] + ld * KVN_S + ln;
        bf16* vd = &Vs[buf][0] + ld * KVN_S + ln;
        async_ld_b128(kd,     kp0 + n0);
        async_ld_b128(kd + 8, kp0 + n0 + 8);
        async_ld_b128(vd,     vp0 + n0);
        async_ld_b128(vd + 8, vp0 + n0 + 8);
    };

    v8f acc = zero8();
    asyncKV(0, 0);
    int cur = 0;
    for (int n0 = 0; n0 < 2048; n0 += 128) {
        const bool more = (n0 + 128) < 2048;
        wait_async0();
        __syncthreads();
        if (more) asyncKV(cur ^ 1, n0 + 128);
#pragma unroll
        for (int s = 0; s < 4; ++s) {
            v16bf a = frag_a (&Ks[cur][0] + ti * KVN_S + s * 32, KVN_S);
            v16bf b = frag_bT(&Vs[cur][0] + tj * KVN_S + s * 32, KVN_S);
            WMMA_BF16(acc, a, b);
        }
        if (more) cur ^= 1;
    }

    const int lrow = (lane >> 4) << 3;
    const int lcol = lane & 15;
    float* kvdst = KV + (size_t)bh * (DHEAD * DHEAD);
#pragma unroll
    for (int r = 0; r < 8; ++r)
        atomicAdd(kvdst + (ti + lrow + r) * DHEAD + (tj + lcol), acc[r]);
}

// ---------------------------------------------------------------------------
// attn: out = (Q @ kv) / (Q . ksum + eps) -> bf16 [B,N,1024]
// ---------------------------------------------------------------------------
__global__ __launch_bounds__(256) void attn_out_kernel(
    const bf16* __restrict__ Q, const float* __restrict__ KV,
    const float* __restrict__ Ksum, bf16* __restrict__ Attn)
{
    __shared__ bf16  kvT[DHEAD * KVT_S];
    __shared__ float ksum_s[DHEAD];
    __shared__ float qks[128];

    const int tid  = threadIdx.x;
    const int bh   = blockIdx.x;
    const int b    = bh >> 4, h = bh & 15;
    const int n0   = blockIdx.y * 128;
    const int wave = tid >> 5;
    const int lane = tid & 31;

    {   // stage kv transposed: thread = (f = tid&63, dg = tid>>6)
        const float* kvsrc = KV + (size_t)bh * (DHEAD * DHEAD);
        const int f = tid & 63, dg = (tid >> 6) << 4;
        v8bf h0, h1;
#pragma unroll
        for (int i = 0; i < 8; ++i) h0[i] = (bf16)kvsrc[(dg + i) * DHEAD + f];
#pragma unroll
        for (int i = 0; i < 8; ++i) h1[i] = (bf16)kvsrc[(dg + 8 + i) * DHEAD + f];
        *(v8bf*)(kvT + f * KVT_S + dg)     = h0;
        *(v8bf*)(kvT + f * KVT_S + dg + 8) = h1;
    }
    if (tid < DHEAD) ksum_s[tid] = Ksum[bh * DHEAD + tid];
    __syncthreads();

    {   // per-row q . ksum: 2 threads/row, vectorized, pair-combined via shfl
        const int qr = tid >> 1;
        const int qh = (tid & 1) << 5;
        const v8bf*   qv  = (const v8bf*)(Q + ((size_t)bh * SEQ + n0 + qr) * DHEAD + qh);
        const float4* ks4 = (const float4*)(ksum_s + qh);
        float p = 0.f;
#pragma unroll
        for (int t = 0; t < 4; ++t) {
            v8bf qq = qv[t];
            float4 k0 = ks4[2 * t], k1 = ks4[2 * t + 1];
            p += (float)qq[0] * k0.x + (float)qq[1] * k0.y +
                 (float)qq[2] * k0.z + (float)qq[3] * k0.w +
                 (float)qq[4] * k1.x + (float)qq[5] * k1.y +
                 (float)qq[6] * k1.z + (float)qq[7] * k1.w;
        }
        p += __shfl_xor(p, 1, 32);
        if ((tid & 1) == 0) qks[qr] = p;
    }
    __syncthreads();

    const size_t qbase = ((size_t)bh * SEQ + n0 + wave * 16) * DHEAD;
    v8f acc[4];
#pragma unroll
    for (int tj = 0; tj < 4; ++tj) acc[tj] = zero8();
#pragma unroll
    for (int s = 0; s < 2; ++s) {
        v16bf a;
        {
            const int row = lane & 15;
            const int kb  = (lane >> 4) << 3;
            const bf16* pq = Q + qbase + (size_t)row * DHEAD + s * 32;
#pragma unroll
            for (int i = 0; i < 8; ++i) {
                a[i]     = pq[kb + i];
                a[i + 8] = pq[16 + kb + i];
            }
        }
#pragma unroll
        for (int tj = 0; tj < 4; ++tj) {
            v16bf bb = frag_bT(kvT + (tj * 16) * KVT_S + s * 32, KVT_S);
            WMMA_BF16(acc[tj], a, bb);
        }
    }

    const int lrow = (lane >> 4) << 3;
    const int lcol = lane & 15;
#pragma unroll
    for (int tj = 0; tj < 4; ++tj)
#pragma unroll
        for (int r = 0; r < 8; ++r) {
            const int rit = wave * 16 + lrow + r;
            const float v = acc[tj][r] / (qks[rit] + EPSV);
            const int n   = n0 + rit;
            const int d   = tj * 16 + lcol;
            Attn[((size_t)b * SEQ + n) * INNER + h * DHEAD + d] = (bf16)v;
        }
}

// ---------------------------------------------------------------------------
// GEMM2: out = attn @ Wout + bout. Both tiles via async-to-LDS (no prefetch
// VGPRs at all); double-buffered. Wave tile 64(M)x32(N).
// ---------------------------------------------------------------------------
__global__ __launch_bounds__(256) void gemm_out(
    const bf16* __restrict__ A,       // [32768,1024] bf16
    const bf16* __restrict__ WT,      // [1024,1024]  bf16 (transposed)
    const float* __restrict__ bias,   // [1024]
    float* __restrict__ Out)          // [32768,1024]
{
    __shared__ bf16 As[2][128 * LDA_S];
    __shared__ bf16 Bt[2][128 * LDBT];

    const int tid     = threadIdx.x;
    const int wave    = tid >> 5;
    const int lane    = tid & 31;
    const int rowBase = blockIdx.y * 128;
    const int colBase = blockIdx.x * 128;
    const int wm      = (wave >> 2) * 64;
    const int wn      = (wave & 3) * 32;

    v8f acc[4][2];
#pragma unroll
    for (int i = 0; i < 4; ++i)
#pragma unroll
        for (int j = 0; j < 2; ++j) acc[i][j] = zero8();

    const int ar = tid >> 1;
    const int ac = (tid & 1) << 4;
    const bf16* arow = A  + (size_t)(rowBase + ar) * INNER + ac;
    const bf16* wrow = WT + (size_t)(colBase + ar) * DIMM + ac;

    auto asyncAB = [&](int buf, int k0) {
        bf16* ad = &As[buf][0] + ar * LDA_S + ac;
        bf16* bd = &Bt[buf][0] + ar * LDBT + ac;
        async_ld_b128(ad,     arow + k0);
        async_ld_b128(ad + 8, arow + k0 + 8);
        async_ld_b128(bd,     wrow + k0);
        async_ld_b128(bd + 8, wrow + k0 + 8);
    };
    auto compute = [&](const bf16* Ab, const bf16* Bb) {
        v16bf bfr[2];
#pragma unroll
        for (int tj = 0; tj < 2; ++tj)
            bfr[tj] = frag_bT(Bb + (wn + tj * 16) * LDBT, LDBT);
#pragma unroll
        for (int ti = 0; ti < 4; ++ti) {
            v16bf a = frag_a(Ab + (wm + ti * 16) * LDA_S, LDA_S);
            WMMA_BF16(acc[ti][0], a, bfr[0]);
            WMMA_BF16(acc[ti][1], a, bfr[1]);
        }
    };

    asyncAB(0, 0);
    int cur = 0;
    for (int k0 = 0; k0 < INNER; k0 += 32) {
        const bool more = (k0 + 32) < INNER;
        wait_async0();
        __syncthreads();
        if (more) asyncAB(cur ^ 1, k0 + 32);
        compute(&As[cur][0], &Bt[cur][0]);
        if (more) cur ^= 1;
    }

    const int lrow = (lane >> 4) << 3;
    const int lcol = lane & 15;
#pragma unroll
    for (int ti = 0; ti < 4; ++ti)
#pragma unroll
        for (int tj = 0; tj < 2; ++tj) {
            const int c = colBase + wn + tj * 16 + lcol;
            const float bs = bias[c];
#pragma unroll
            for (int r = 0; r < 8; ++r) {
                const int m = rowBase + wm + ti * 16 + lrow + r;
                Out[(size_t)m * DIMM + c] = acc[ti][tj][r] + bs;
            }
        }
}

// ---------------------------------------------------------------------------
extern "C" void kernel_launch(void* const* d_in, const int* in_sizes, int n_in,
                              void* d_out, int out_size, void* d_ws, size_t ws_size,
                              hipStream_t stream) {
    const float* x    = (const float*)d_in[0];
    const float* Wqkv = (const float*)d_in[1];
    const float* Wout = (const float*)d_in[2];
    const float* bout = (const float*)d_in[3];

    char* ws = (char*)d_ws;
    size_t off = 0;
    auto alloc = [&](size_t bytes) -> void* {
        void* p = ws + off;
        off = (off + bytes + 255) & ~(size_t)255;
        return p;
    };

    bf16* wqkvT  = (bf16*)alloc((size_t)TRIPLE * DIMM * 2);   // [3072][1024]
    bf16* woutT  = (bf16*)alloc((size_t)DIMM * INNER * 2);    // [1024][1024]
    bf16* Qb     = (bf16*)alloc((size_t)BATCH * HEADS * SEQ * DHEAD * 2);  // [bh][n][d]
    bf16* KTb    = (bf16*)alloc((size_t)BATCH * HEADS * SEQ * DHEAD * 2);  // [bh][d][n]
    bf16* VTb    = (bf16*)alloc((size_t)BATCH * HEADS * SEQ * DHEAD * 2);  // [bh][d][n]
    bf16* Attn   = (bf16*)alloc((size_t)MROWS * INNER * 2);
    float* KVbuf = (float*)alloc((size_t)BATCH * HEADS * DHEAD * DHEAD * 4);
    float* KSbuf = (float*)alloc((size_t)BATCH * HEADS * DHEAD * 4);

    cvt_transpose<<<dim3(TRIPLE / 32, DIMM / 32), 256, 0, stream>>>(Wqkv, wqkvT, DIMM, TRIPLE);
    cvt_transpose<<<dim3(DIMM / 32, INNER / 32), 256, 0, stream>>>(Wout, woutT, INNER, DIMM);

    const int nKV = BATCH * HEADS * DHEAD * DHEAD;   // 262144
    zero_f32<<<(nKV + 255) / 256, 256, 0, stream>>>(KVbuf, nKV);

    gemm_qkv<<<dim3(TRIPLE / 128, MROWS / 128), 256, 0, stream>>>(x, wqkvT, Qb, KTb, VTb);
    ksum_kernel<<<BATCH * HEADS * DHEAD, 256, 0, stream>>>(KTb, KSbuf);
    kv_kernel<<<dim3(BATCH * HEADS, 4), 512, 0, stream>>>(KTb, VTb, KVbuf);
    attn_out_kernel<<<dim3(BATCH * HEADS, SEQ / 128), 256, 0, stream>>>(Qb, KVbuf, KSbuf, Attn);
    gemm_out<<<dim3(DIMM / 128, MROWS / 128), 256, 0, stream>>>(Attn, woutT, bout, (float*)d_out);
}
